// LSTM_64896955843191
// MI455X (gfx1250) — matrix operand order repaired and checked by reading, not compiled, not run on hardware
//
#include <hip/hip_runtime.h>

// ---------------------------------------------------------------------------
// LSTM forward for MI455X (gfx1250), wave32 + v_wmma_f32_16x16x32_bf16.
//
// gates_t = x_t @ Wi + h_{t-1} @ Wh + b   (K = 512 + 1024 = 1536, Ncols = 4096)
// c_t = sigm(i)*tanh(g) + sigm(f)*c_{t-1};  h_t = sigm(o)*tanh(c_t)
//
// One wave owns one 16x16 (batch-tile, hidden-col-tile) block and its 4 gate
// accumulators. All 8 waves of a block share the same batch tile, so the
// block's A rows (x_t and h_{t-1}) are staged in LDS once and re-read from
// there (bank-conflict-free via row padding). The 4 gate weight streams share
// one base pointer; the constant gate stride (1 MB / 2 MB) folds into the
// global_load_b128 immediate offset.
// ---------------------------------------------------------------------------

#define NB   256   // batch
#define SS   512   // timesteps
#define FF   512   // input features
#define HH   1024  // hidden

#define AXP  (FF + 8)    // padded LDS row (halfs): 260 dwords == 4 mod 64 banks
#define AHP  (HH + 8)    // padded LDS row (halfs): 516 dwords == 4 mod 64 banks

#define GSTRIDE_WI (64 * 16 * 512)   // halfs between gates in packed Wi (1 MB)
#define GSTRIDE_WH (64 * 32 * 512)   // halfs between gates in packed Wh (2 MB)

typedef unsigned short u16;
typedef __attribute__((ext_vector_type(16))) __bf16 v16bf;
typedef __attribute__((ext_vector_type(8)))  float  v8f;

union V16 { uint4 q[2]; v16bf v; };

__device__ __forceinline__ u16 f2bf(float f) {
    unsigned u = __builtin_bit_cast(unsigned, f);
    u += 0x7fffu + ((u >> 16) & 1u);           // round-to-nearest-even
    return (u16)(u >> 16);
}
__device__ __forceinline__ float sigm(float x) { return 1.f / (1.f + __expf(-x)); }
__device__ __forceinline__ float tanh_fast(float x) {
    float e = __expf(-2.f * x);
    return (1.f - e) / (1.f + e);
}

// ---------------------------------------------------------------------------
// Pack one gate weight matrix W [K x 1024] (fp32, row-major) into bf16 tiles
// in this kernel's WMMA B-operand order, contiguous K per lane half (32x16 per tile):
//   tile (nt4 = gate*64 + jt, kt): 32 lanes x 16 halfs, lane-contiguous 32B.
//   lane<16 : N = lane,    K = kt*32 + 0..15
//   lane>=16: N = lane-16, K = kt*32 + 16..31
// ---------------------------------------------------------------------------
__global__ void k_pack_w(const float* __restrict__ W, u16* __restrict__ out,
                         int KT, int gate) {
    int tid = blockIdx.x * blockDim.x + threadIdx.x;
    int lane = tid & 31;
    int kt   = (tid >> 5) % KT;
    int jt   = (tid >> 5) / KT;
    if (jt >= 64) return;
    int j     = jt * 16 + (lane & 15);
    int kbase = kt * 32 + ((lane < 16) ? 0 : 16);
    alignas(16) u16 tmp[16];
#pragma unroll
    for (int i = 0; i < 16; ++i)
        tmp[i] = f2bf(W[(size_t)(kbase + i) * HH + j]);
    size_t o = ((size_t)((gate * 64 + jt) * KT + kt) * 32 + lane) * 16;
    uint4* dst = (uint4*)(out + o);
    dst[0] = *(uint4*)&tmp[0];
    dst[1] = *(uint4*)&tmp[8];
}

// x fp32 [N][S][F] -> bf16, same ordering (row-major rows are the WMMA A rows)
__global__ void k_conv_x(const float4* __restrict__ x, u16* __restrict__ out) {
    size_t i = (size_t)blockIdx.x * blockDim.x + threadIdx.x;   // 4 elems each
    float4 v = x[i];
    alignas(8) u16 t[4] = { f2bf(v.x), f2bf(v.y), f2bf(v.z), f2bf(v.w) };
    ((uint2*)out)[i] = *(uint2*)t;
}

// zero c0 (fp32) and h0 (bf16)
__global__ void k_init(float* __restrict__ c, unsigned* __restrict__ hb) {
    int i = blockIdx.x * blockDim.x + threadIdx.x;
    if (i < NB * HH) c[i] = 0.f;
    if (i < NB * HH / 2) hb[i] = 0u;
}

// ---------------------------------------------------------------------------
// One LSTM timestep. Grid = 1024 waves (128 blocks x 256 thr = 8 waves/blk).
// block b: ntile = b/8; wave w in block: jt = (b%8)*8 + w.
// ---------------------------------------------------------------------------
__global__ __launch_bounds__(256) void k_step(
    const u16* __restrict__ xb,   const u16* __restrict__ WiP,
    const u16* __restrict__ WhP,
    const float* __restrict__ bI, const float* __restrict__ bF,
    const float* __restrict__ bG, const float* __restrict__ bO,
    const u16* __restrict__ hin,  u16* __restrict__ hout,
    float* __restrict__ cbuf,
    float* __restrict__ hFinal,   float* __restrict__ cFinal,
    int t, int last)
{
    __shared__ u16 Ax[16 * AXP];   // x_t rows for this batch tile
    __shared__ u16 Ah[16 * AHP];   // h_{t-1} rows for this batch tile

    const int tid   = threadIdx.x;
    const int lane  = tid & 31;
    const int wave  = (blockIdx.x << 3) | (tid >> 5);
    const int ntile = blockIdx.x >> 3;  // 0..15 (shared by all waves in block)
    const int jt    = wave & 63;        // 0..63
    const int mrow  = lane & 15;

    // ---- cooperative stage of A rows into LDS ----
    // x: 16 rows x 64 chunks of 8 halfs = 1024 chunks
    // h: 16 rows x 128 chunks           = 2048 chunks   (3072 total, 12/thread)
#pragma unroll
    for (int i = 0; i < 12; ++i) {
        int chunk = tid + i * 256;
        if (chunk < 1024) {
            int row = chunk >> 6, c = chunk & 63;
            int n = ntile * 16 + row;
            uint4 v = *(const uint4*)(xb + ((size_t)n * SS + t) * FF + c * 8);
            *(uint4*)&Ax[row * AXP + c * 8] = v;
        } else {
            int ch = chunk - 1024;
            int row = ch >> 7, c = ch & 127;
            int n = ntile * 16 + row;
            uint4 v = *(const uint4*)(hin + (size_t)n * HH + c * 8);
            *(uint4*)&Ah[row * AHP + c * 8] = v;
        }
    }
    __syncthreads();

    const int aoff = (lane < 16) ? 0 : 16;         // A K offset (contiguous order, matches B packing)
    const u16* axl = &Ax[mrow * AXP + aoff];
    const u16* ahl = &Ah[mrow * AHP + aoff];

    v8f acc[4] = {};    // i, f, g, o accumulators for this 16x16 block

    // Single per-wave B base per weight matrix; gate offset is a constant
    // fold-able immediate (Wi: 1 MB stride, Wh: 2 MB stride, both < 8 MB).
    const u16* bpx = WiP + (size_t)jt * 16 * 512 + (size_t)lane * 16;
    const u16* bph = WhP + (size_t)jt * 32 * 512 + (size_t)lane * 16;

    // ---- input projection: K = 512 over x_t ----
#pragma unroll 2
    for (int kt = 0; kt < FF / 32; ++kt) {
        V16 A;
        A.q[0] = *(const uint4*)(axl + kt * 32);
        A.q[1] = *(const uint4*)(axl + kt * 32 + 8);
        const u16* bp = bpx + (size_t)kt * 512;
#pragma unroll
        for (int g = 0; g < 4; ++g) {
            const u16* bg_ = bp + (size_t)g * GSTRIDE_WI;
            V16 B;
            B.q[0] = *(const uint4*)(bg_);
            B.q[1] = *(const uint4*)(bg_ + 8);
            acc[g] = __builtin_amdgcn_wmma_f32_16x16x32_bf16(
                false, A.v, false, B.v, (short)0, acc[g], false, false);
        }
    }

    // ---- recurrent projection: K = 1024 over h_{t-1} ----
#pragma unroll 2
    for (int kt = 0; kt < HH / 32; ++kt) {
        V16 A;
        A.q[0] = *(const uint4*)(ahl + kt * 32);
        A.q[1] = *(const uint4*)(ahl + kt * 32 + 8);
        const u16* bp = bph + (size_t)kt * 512;
#pragma unroll
        for (int g = 0; g < 4; ++g) {
            const u16* bg_ = bp + (size_t)g * GSTRIDE_WH;
            V16 B;
            B.q[0] = *(const uint4*)(bg_);
            B.q[1] = *(const uint4*)(bg_ + 8);
            acc[g] = __builtin_amdgcn_wmma_f32_16x16x32_bf16(
                false, A.v, false, B.v, (short)0, acc[g], false, false);
        }
    }

    // ---- epilogue: bias, activations, state update ----
    // C/D layout: VGPR r holds (M = r + 8*(lane>=16), N = lane&15).
    const int j  = jt * 16 + mrow;
    const int hi = lane >> 4;
    const float Bi = bI[j], Bf = bF[j], Bg = bG[j], Bo = bO[j];
#pragma unroll
    for (int r = 0; r < 8; ++r) {
        size_t idx = (size_t)(ntile * 16 + r + hi * 8) * HH + j;
        float iv = sigm(acc[0][r] + Bi);
        float fv = sigm(acc[1][r] + Bf);
        float gv = tanh_fast(acc[2][r] + Bg);
        float ov = sigm(acc[3][r] + Bo);
        float cp = cbuf[idx];
        float cn = iv * gv + fv * cp;
        float hn = ov * tanh_fast(cn);
        cbuf[idx] = cn;                 // each element owned by exactly one lane
        hout[idx] = f2bf(hn);           // bf16 h for next step's A operand
        if (last) { hFinal[idx] = hn; cFinal[idx] = cn; }
    }
}

// ---------------------------------------------------------------------------
// Host launcher. d_in order: x, Wii, Whi, bi, Wif, Whf, bf, Wig, Whg, bg,
// Wio, Who, bo. d_out = [h (256x1024) fp32, c (256x1024) fp32].
// ---------------------------------------------------------------------------
extern "C" void kernel_launch(void* const* d_in, const int* in_sizes, int n_in,
                              void* d_out, int out_size, void* d_ws, size_t ws_size,
                              hipStream_t stream) {
    const float* x   = (const float*)d_in[0];
    const float* Wii = (const float*)d_in[1];
    const float* Whi = (const float*)d_in[2];
    const float* bi  = (const float*)d_in[3];
    const float* Wif = (const float*)d_in[4];
    const float* Whf = (const float*)d_in[5];
    const float* bf  = (const float*)d_in[6];
    const float* Wig = (const float*)d_in[7];
    const float* Whg = (const float*)d_in[8];
    const float* bg  = (const float*)d_in[9];
    const float* Wio = (const float*)d_in[10];
    const float* Who = (const float*)d_in[11];
    const float* bo  = (const float*)d_in[12];

    // workspace layout (bytes)
    char* ws = (char*)d_ws;
    u16*   WiP  = (u16*)(ws);                                  //   4 MB
    u16*   WhP  = (u16*)(ws + (4ull  << 20));                  //   8 MB
    u16*   xb   = (u16*)(ws + (12ull << 20));                  // 128 MB
    float* cbuf = (float*)(ws + (140ull << 20));               //   1 MB
    u16*   hb0  = (u16*)(ws + (141ull << 20));                 // 0.5 MB
    u16*   hb1  = hb0 + NB * HH;                               // 0.5 MB

    // pack weights (i, f, g, o)
    k_pack_w<<<128, 256, 0, stream>>>(Wii, WiP, 16, 0);
    k_pack_w<<<128, 256, 0, stream>>>(Wif, WiP, 16, 1);
    k_pack_w<<<128, 256, 0, stream>>>(Wig, WiP, 16, 2);
    k_pack_w<<<128, 256, 0, stream>>>(Wio, WiP, 16, 3);
    k_pack_w<<<256, 256, 0, stream>>>(Whi, WhP, 32, 0);
    k_pack_w<<<256, 256, 0, stream>>>(Whf, WhP, 32, 1);
    k_pack_w<<<256, 256, 0, stream>>>(Whg, WhP, 32, 2);
    k_pack_w<<<256, 256, 0, stream>>>(Who, WhP, 32, 3);

    // convert x to bf16 (4 floats / thread)
    k_conv_x<<<(NB * SS * FF / 4) / 256, 256, 0, stream>>>((const float4*)x, xb);

    // zero c0 and h0
    k_init<<<(NB * HH) / 256, 256, 0, stream>>>(cbuf, (unsigned*)hb0);

    float* outH = (float*)d_out;
    float* outC = outH + NB * HH;

    for (int t = 0; t < SS; ++t) {
        const u16* hin = (t & 1) ? hb1 : hb0;
        u16*      hout = (t & 1) ? hb0 : hb1;
        k_step<<<128, 256, 0, stream>>>(xb, WiP, WhP, bi, bf, bg, bo,
                                        hin, hout, cbuf, outH, outC,
                                        t, (t == SS - 1) ? 1 : 0);
    }
}
